// StageBlock_53712861003949
// MI455X (gfx1250) — matrix operand order repaired and checked by reading, not compiled, not run on hardware
//
#include <hip/hip_runtime.h>
#include <hip/hip_bf16.h>

typedef __attribute__((ext_vector_type(16))) _Float16 v16h;
typedef __attribute__((ext_vector_type(8)))  float    v8f;
typedef __attribute__((ext_vector_type(2)))  _Float16 h2v;

#define EPSV   1e-5f
#define KNN_K  16
#define NBATCH 128
#define NPT    128
#define MTOT   (NBATCH*NPT*KNN_K)   /* 262144 edge-conv pixels */
#define M2TOT  (NBATCH*NPT)         /* 16384 shortcut pixels   */

// ---------------------------------------------------------------- mask + prep
__global__ void k_mask(const float* __restrict__ x, const float* __restrict__ pts,
                       float* __restrict__ xm, float* __restrict__ pm,
                       float* __restrict__ maskf, float* __restrict__ counts) {
    __shared__ float red[NPT];
    int b = blockIdx.x, n = threadIdx.x;
    float s = 0.f;
    for (int c = 0; c < 7; ++c) s += fabsf(x[(b*7 + c)*NPT + n]);
    float mf = (s != 0.f) ? 1.f : 0.f;
    maskf[b*NPT + n] = mf;
    for (int c = 0; c < 7; ++c) xm[(b*7 + c)*NPT + n] = x[(b*7 + c)*NPT + n] * mf;
    for (int c = 0; c < 2; ++c) pm[(b*2 + c)*NPT + n] = pts[(b*2 + c)*NPT + n] * mf;
    red[n] = mf;
    __syncthreads();
    for (int off = 64; off > 0; off >>= 1) {
        if (n < off) red[n] += red[n + off];
        __syncthreads();
    }
    if (n == 0) counts[b] = fmaxf(red[0], 1.f);
}

__global__ void k_init_ident(_Float16* ones_h, _Float16* zeros_h) {
    int t = threadIdx.x;
    ones_h[t]  = (_Float16)1.f;
    zeros_h[t] = (_Float16)0.f;
}

// -------------------------------------------------- per-channel sum / sumsq
__global__ void k_stats_f32(const float* __restrict__ X, long strideO, long strideB,
                            int Nn, long total, float* sums, float* sumsq) {
    __shared__ float rs[256], rq[256];
    int o = blockIdx.x, t = threadIdx.x;
    float s = 0.f, q = 0.f;
    for (long j = t; j < total; j += 256) {
        long bI = j / Nn, n = j % Nn;
        float v = X[(long)o*strideO + bI*strideB + n];
        s += v; q += v*v;
    }
    rs[t] = s; rq[t] = q; __syncthreads();
    for (int off = 128; off > 0; off >>= 1) {
        if (t < off) { rs[t] += rs[t+off]; rq[t] += rq[t+off]; }
        __syncthreads();
    }
    if (t == 0) { sums[o] = rs[0]; sumsq[o] = rq[0]; }
}

// paired f16 stats: one block per channel pair, both channels at once
__global__ void k_stats_pair(const h2v* __restrict__ Xp, long M,
                             float* sums, float* sumsq) {
    __shared__ float r0[256], q0[256], r1[256], q1[256];
    int p = blockIdx.x, t = threadIdx.x;
    float s0 = 0.f, a0 = 0.f, s1 = 0.f, a1 = 0.f;
    for (long j = t; j < M; j += 256) {
        h2v v = Xp[(long)p*M + j];
        float x0 = (float)v[0], x1 = (float)v[1];
        s0 += x0; a0 += x0*x0; s1 += x1; a1 += x1*x1;
    }
    r0[t] = s0; q0[t] = a0; r1[t] = s1; q1[t] = a1; __syncthreads();
    for (int off = 128; off > 0; off >>= 1) {
        if (t < off) {
            r0[t] += r0[t+off]; q0[t] += q0[t+off];
            r1[t] += r1[t+off]; q1[t] += q1[t+off];
        }
        __syncthreads();
    }
    if (t == 0) {
        sums[2*p] = r0[0]; sumsq[2*p] = q0[0];
        sums[2*p+1] = r1[0]; sumsq[2*p+1] = q1[0];
    }
}

__global__ void k_bn_finalize(const float* sums, const float* sumsq,
                              const float* g, const float* bb, float invCount,
                              float* scale, float* bias,
                              _Float16* scale_h, _Float16* bias_h) {
    int o = threadIdx.x;
    float mean = sums[o] * invCount;
    float var  = sumsq[o] * invCount - mean*mean;
    float sc   = g[o] * rsqrtf(var + EPSV);
    float bi   = bb[o] - mean*sc;
    scale[o] = sc;          bias[o] = bi;
    scale_h[o] = (_Float16)sc; bias_h[o] = (_Float16)bi;
}

__global__ void k_apply_bn_mask(float* __restrict__ x, const float* scale,
                                const float* bias, const float* maskf, int C) {
    int t = blockIdx.x*256 + threadIdx.x;        // t == ((b*C)+c)*NPT + n
    int n = t % NPT;
    int c = (t / NPT) % C;
    int b = t / (NPT*C);
    x[t] = (scale[c]*x[t] + bias[c]) * maskf[b*NPT + n];
}

// ---------------------------------------------------------------- KNN (k=16)
__global__ __launch_bounds__(256)
void k_knn(const float* __restrict__ P, int D,
           const float* __restrict__ maskf, int* __restrict__ idxout) {
    int wave = threadIdx.x >> 5, lane = threadIdx.x & 31;
    int q = blockIdx.x*8 + wave;
    int b = q / NPT, n = q % NPT;
    float shn = (1.f - maskf[b*NPT + n]) * 1e9f;
    float d2[4], shm[4];
    for (int j = 0; j < 4; ++j) {
        int m = lane + 32*j;
        shm[j] = (1.f - maskf[b*NPT + m]) * 1e9f;
        d2[j] = 0.f;
    }
    for (int c = 0; c < D; ++c) {
        float pn = P[((long)b*D + c)*NPT + n] + shn;
        for (int j = 0; j < 4; ++j) {
            int m = lane + 32*j;
            float pm = P[((long)b*D + c)*NPT + m] + shm[j];
            float df = pn - pm;
            d2[j] += df*df;
        }
    }
    for (int t = 0; t <= KNN_K; ++t) {
        float bv = 3.4e38f; int bi = 0x7fffffff;
        for (int j = 0; j < 4; ++j) {
            int m = lane + 32*j;
            if (d2[j] < bv || (d2[j] == bv && m < bi)) { bv = d2[j]; bi = m; }
        }
        for (int off = 16; off > 0; off >>= 1) {
            float ov = __shfl_xor(bv, off, 32);
            int   oi = __shfl_xor(bi, off, 32);
            if (ov < bv || (ov == bv && oi < bi)) { bv = ov; bi = oi; }
        }
        if (t > 0 && lane == 0) idxout[(b*NPT + n)*KNN_K + (t-1)] = bi;
        if ((bi & 31) == lane) d2[bi >> 5] = 3.4e38f;
    }
}

// ------------------------------- weight pre-pack into WMMA A-fragment layout
// Af[(ot*nsteps + s)*32 + lane] = v16h fragment (zero-padded past Creal)
__global__ void k_pack_w(const float* __restrict__ W, int Creal, int nsteps, int O,
                         v16h* __restrict__ Af) {
    int t = blockIdx.x*256 + threadIdx.x;
    int lane = t & 31;
    int idx  = t >> 5;
    int total = (O >> 4) * nsteps;
    if (idx >= total) return;
    int s  = idx % nsteps, ot = idx / nsteps;
    int c0 = s << 5;
    int hlf = lane >> 4, mrow = lane & 15;
    int o = (ot << 4) + mrow;
    v16h a;
#pragma unroll
    for (int i = 0; i < 8; ++i) {
        int kb = 16*hlf + 2*i;
        int c  = c0 + kb;
        a[2*i]   = (c   < Creal) ? (_Float16)W[(long)o*Creal + c]   : (_Float16)0.f;
        a[2*i+1] = (c+1 < Creal) ? (_Float16)W[(long)o*Creal + c+1] : (_Float16)0.f;
    }
    Af[idx*32 + lane] = a;
}

// ---------------------- graph feature pack -> paired f16 [Cpad/2][MTOT] h2v
__device__ __forceinline__ float gf_val(const float* x, int cin, int c,
                                        int b, int n, int nbr) {
    if (c < cin)   return x[((long)b*cin + c)*NPT + n];
    if (c < 2*cin) return x[((long)b*cin + c-cin)*NPT + nbr]
                        - x[((long)b*cin + c-cin)*NPT + n];
    return 0.f;
}
__global__ void k_pack_graph(const float* __restrict__ x, int cin, int Cpad,
                             const int* __restrict__ idx, h2v* __restrict__ Hp) {
    long m = (long)blockIdx.x*256 + threadIdx.x;     // < MTOT
    int b = (int)(m >> 11);                          // N*K = 2048
    int r = (int)(m & 2047);
    int n = r >> 4, k = r & 15;
    int nbr = idx[(b*NPT + n)*KNN_K + k];
    int np = Cpad >> 1;
    for (int p = 0; p < np; ++p) {
        h2v h;
        h[0] = (_Float16)gf_val(x, cin, 2*p,   b, n, nbr);
        h[1] = (_Float16)gf_val(x, cin, 2*p+1, b, n, nbr);
        Hp[(long)p*MTOT + m] = h;
    }
}

// ------------- shortcut input pack -> paired f16 [Cpad/2][M2TOT] (zero pad)
__global__ void k_pack_xsc(const float* __restrict__ x, int cin, int Cpad,
                           h2v* __restrict__ Hp) {
    int m = blockIdx.x*256 + threadIdx.x;            // < M2TOT
    int b = m >> 7, n = m & 127;
    for (int p = 0; p < (Cpad >> 1); ++p) {
        int c0 = 2*p, c1 = c0 + 1;
        h2v h;
        h[0] = (_Float16)((c0 < cin) ? x[((long)b*cin + c0)*NPT + n] : 0.f);
        h[1] = (_Float16)((c1 < cin) ? x[((long)b*cin + c1)*NPT + n] : 0.f);
        Hp[(long)p*M2TOT + m] = h;
    }
}

// -------------------------------------------------------- core WMMA GEMM
// Each wave: 4 o-tiles (64 out channels) x 16 pixels.  Per K=32 step:
// one BN'd B fragment (8 coalesced b32 loads + v_pk_fma/v_pk_max) feeds
// FOUR independent v_wmma_f32_16x16x32_f16 (prepacked A fragments).
__global__ __launch_bounds__(256)
void k_wmma_gemm(const v16h* __restrict__ Af, int nsteps,
                 const h2v* __restrict__ Hp,
                 const h2v* __restrict__ sc2, const h2v* __restrict__ bi2,
                 h2v* __restrict__ Outp, int O, long M) {
    int ogroups = O >> 6;                            // groups of 4 o-tiles
    int og  = blockIdx.x % ogroups;
    int mt8 = blockIdx.x / ogroups;
    int lane = threadIdx.x & 31, wave = threadIdx.x >> 5;
    int hlf = lane >> 4, mrow = lane & 15;
    long m = (long)(mt8*8 + wave)*16 + mrow;
    const v16h* ap = Af + (size_t)(og*4*nsteps)*32 + lane;
    const h2v zero2 = {(_Float16)0.f, (_Float16)0.f};
    v8f acc0 = {0.f,0.f,0.f,0.f,0.f,0.f,0.f,0.f};
    v8f acc1 = acc0, acc2 = acc0, acc3 = acc0;
    for (int s = 0; s < nsteps; ++s) {
        v16h bf;
        int p0 = s*16 + 8*hlf;
#pragma unroll
        for (int i = 0; i < 8; ++i) {
            int p = p0 + i;
            h2v v = Hp[(long)p*M + m];
            v = __builtin_elementwise_max(sc2[p]*v + bi2[p], zero2);
            ((h2v*)&bf)[i] = v;
        }
        v16h a0 = ap[(size_t)(0*nsteps + s)*32];
        v16h a1 = ap[(size_t)(1*nsteps + s)*32];
        v16h a2 = ap[(size_t)(2*nsteps + s)*32];
        v16h a3 = ap[(size_t)(3*nsteps + s)*32];
        acc0 = __builtin_amdgcn_wmma_f32_16x16x32_f16(false, a0, false, bf,
                                                      (short)0, acc0, false, false);
        acc1 = __builtin_amdgcn_wmma_f32_16x16x32_f16(false, a1, false, bf,
                                                      (short)0, acc1, false, false);
        acc2 = __builtin_amdgcn_wmma_f32_16x16x32_f16(false, a2, false, bf,
                                                      (short)0, acc2, false, false);
        acc3 = __builtin_amdgcn_wmma_f32_16x16x32_f16(false, a3, false, bf,
                                                      (short)0, acc3, false, false);
    }
    v8f accs[4] = {acc0, acc1, acc2, acc3};
#pragma unroll
    for (int tIdx = 0; tIdx < 4; ++tIdx) {
        int pbase = ((og*4 + tIdx) << 3) + 4*hlf;    // output channel-pair base
#pragma unroll
        for (int j = 0; j < 4; ++j) {
            h2v o2;
            o2[0] = (_Float16)accs[tIdx][2*j];
            o2[1] = (_Float16)accs[tIdx][2*j+1];
            Outp[(long)(pbase + j)*M + m] = o2;
        }
    }
}

// shortcut GEMM: same 4-tile core, no input BN, f32 output [O][M]
__global__ __launch_bounds__(256)
void k_wmma_gemm_f32out(const v16h* __restrict__ Af, int nsteps,
                        const h2v* __restrict__ Hp, float* __restrict__ Out,
                        int O, long M) {
    int ogroups = O >> 6;
    int og  = blockIdx.x % ogroups;
    int mt8 = blockIdx.x / ogroups;
    int lane = threadIdx.x & 31, wave = threadIdx.x >> 5;
    int hlf = lane >> 4, mrow = lane & 15;
    long m = (long)(mt8*8 + wave)*16 + mrow;
    const v16h* ap = Af + (size_t)(og*4*nsteps)*32 + lane;
    v8f acc0 = {0.f,0.f,0.f,0.f,0.f,0.f,0.f,0.f};
    v8f acc1 = acc0, acc2 = acc0, acc3 = acc0;
    for (int s = 0; s < nsteps; ++s) {
        v16h bf;
        int p0 = s*16 + 8*hlf;
#pragma unroll
        for (int i = 0; i < 8; ++i)
            ((h2v*)&bf)[i] = Hp[(long)(p0 + i)*M + m];
        v16h a0 = ap[(size_t)(0*nsteps + s)*32];
        v16h a1 = ap[(size_t)(1*nsteps + s)*32];
        v16h a2 = ap[(size_t)(2*nsteps + s)*32];
        v16h a3 = ap[(size_t)(3*nsteps + s)*32];
        acc0 = __builtin_amdgcn_wmma_f32_16x16x32_f16(false, a0, false, bf,
                                                      (short)0, acc0, false, false);
        acc1 = __builtin_amdgcn_wmma_f32_16x16x32_f16(false, a1, false, bf,
                                                      (short)0, acc1, false, false);
        acc2 = __builtin_amdgcn_wmma_f32_16x16x32_f16(false, a2, false, bf,
                                                      (short)0, acc2, false, false);
        acc3 = __builtin_amdgcn_wmma_f32_16x16x32_f16(false, a3, false, bf,
                                                      (short)0, acc3, false, false);
    }
    v8f accs[4] = {acc0, acc1, acc2, acc3};
#pragma unroll
    for (int tIdx = 0; tIdx < 4; ++tIdx) {
        int o_base = (og*4 + tIdx) << 4;
#pragma unroll
        for (int r = 0; r < 8; ++r)
            Out[(long)(o_base + r + 8*hlf)*M + m] = accs[tIdx][r];
    }
}

// -------- epilogue: relu(bn(sc) + mean_k relu(bn(h))) * mask; 2 channels/thr
__global__ void k_epilogue(const h2v* __restrict__ Hrawp, const float* hs, const float* hb,
                           const float* __restrict__ SC, const float* ss, const float* sb,
                           const float* __restrict__ maskf, float* __restrict__ xout, int O) {
    long t = (long)blockIdx.x*256 + threadIdx.x;     // enumerates [B][O/2][N]
    int half_O = O >> 1;
    int n = (int)(t % NPT);
    int p = (int)((t / NPT) % half_O);
    int b = (int)(t / ((long)NPT*half_O));
    long mb = ((long)b*NPT + n)*KNN_K;
    int o0 = 2*p, o1 = o0 + 1;
    float hs0 = hs[o0], hb0 = hb[o0], hs1 = hs[o1], hb1 = hb[o1];
    float f0 = 0.f, f1 = 0.f;
    for (int k = 0; k < KNN_K; ++k) {
        h2v v = Hrawp[(long)p*MTOT + mb + k];
        f0 += fmaxf(hs0*(float)v[0] + hb0, 0.f);
        f1 += fmaxf(hs1*(float)v[1] + hb1, 0.f);
    }
    f0 *= (1.f/KNN_K); f1 *= (1.f/KNN_K);
    float mk = maskf[b*NPT + n];
    float s0 = ss[o0]*SC[(long)o0*M2TOT + b*NPT + n] + sb[o0];
    float s1 = ss[o1]*SC[(long)o1*M2TOT + b*NPT + n] + sb[o1];
    xout[((long)b*O + o0)*NPT + n] = fmaxf(s0 + f0, 0.f) * mk;
    xout[((long)b*O + o1)*NPT + n] = fmaxf(s1 + f1, 0.f) * mk;
}

// ---------------------------------------------------------------- tail MLP
__global__ void k_pool(const float* __restrict__ x, const float* counts, float* xm) {
    int b = blockIdx.x, o = threadIdx.x;
    float s = 0.f;
    for (int n = 0; n < NPT; ++n) s += x[((long)b*256 + o)*NPT + n];
    xm[b*256 + o] = s / counts[b];
}
__global__ void k_mlp1(const float* xm, const float* W1, const float* b1, float* h) {
    int b = blockIdx.x, o = threadIdx.x;
    float s = b1[o];
    for (int i = 0; i < 256; ++i) s += xm[b*256 + i]*W1[o*256 + i];
    h[b*256 + o] = fmaxf(s, 0.f);
}
__global__ void k_mlp2(const float* h, const float* W2, const float* b2, float* out) {
    int t = threadIdx.x, b = t >> 1, j = t & 1;
    float s = b2[j];
    for (int i = 0; i < 256; ++i) s += h[b*256 + i]*W2[j*256 + i];
    out[b*2 + j] = s;
}

// ---------------------------------------------------------------- launcher
extern "C" void kernel_launch(void* const* d_in, const int* in_sizes, int n_in,
                              void* d_out, int out_size, void* d_ws, size_t ws_size,
                              hipStream_t stream) {
    (void)in_sizes; (void)n_in; (void)out_size; (void)ws_size;
    const float* x_in  = (const float*)d_in[0];
    const float* p_in  = (const float*)d_in[1];
    const float* fts_b = (const float*)d_in[2];
    const float* fts_g = (const float*)d_in[3];
    const float *bnb[3][3], *bng[3][3], *Wc[3][3], *scbnb[3], *scbng[3], *scw[3];
    int base = 4;
    for (int i = 0; i < 3; ++i) {
        for (int j = 0; j < 3; ++j) bnb[i][j] = (const float*)d_in[base + j];
        for (int j = 0; j < 3; ++j) bng[i][j] = (const float*)d_in[base + 3 + j];
        for (int j = 0; j < 3; ++j) Wc[i][j]  = (const float*)d_in[base + 6 + j];
        scbnb[i] = (const float*)d_in[base + 9];
        scbng[i] = (const float*)d_in[base + 10];
        scw[i]   = (const float*)d_in[base + 11];
        base += 12;
    }
    const float* W1 = (const float*)d_in[base + 0];
    const float* W2 = (const float*)d_in[base + 1];
    const float* b1 = (const float*)d_in[base + 2];
    const float* b2 = (const float*)d_in[base + 3];

    char* w = (char*)d_ws;
    auto carve = [&](size_t bytes) -> void* {
        void* p = (void*)w; w += (bytes + 255) & ~(size_t)255; return p;
    };
    float*     xbuf0 = (float*)    carve((size_t)NBATCH*256*NPT*4);
    float*     xbuf1 = (float*)    carve((size_t)NBATCH*256*NPT*4);
    float*     pbuf  = (float*)    carve((size_t)NBATCH*2*NPT*4);
    float*     maskf = (float*)    carve((size_t)NBATCH*NPT*4);
    float*     cnts  = (float*)    carve(512);
    int*       idxb  = (int*)      carve((size_t)MTOT*4);
    h2v*       P0    = (h2v*)      carve((size_t)128*MTOT*4);   // 128 pairs
    h2v*       P1    = (h2v*)      carve((size_t)128*MTOT*4);
    h2v*       Hsc   = (h2v*)      carve((size_t)64*M2TOT*4);   // 64 pairs max
    v16h*      Af    = (v16h*)     carve((size_t)16*8*32*32);   // 128 KiB
    float*     scb   = (float*)    carve((size_t)256*M2TOT*4);
    float*     sums  = (float*)    carve(1024);
    float*     sumsq = (float*)    carve(1024);
    float*     s_hs  = (float*)    carve(1024);
    float*     s_hb  = (float*)    carve(1024);
    float*     s_ss  = (float*)    carve(1024);
    float*     s_sb  = (float*)    carve(1024);
    _Float16*  s_hsh = (_Float16*) carve(512);
    _Float16*  s_hbh = (_Float16*) carve(512);
    _Float16*  s_ssh = (_Float16*) carve(512);
    _Float16*  s_sbh = (_Float16*) carve(512);
    _Float16*  id1h  = (_Float16*) carve(512);
    _Float16*  id0h  = (_Float16*) carve(512);
    float*     xmp   = (float*)    carve((size_t)NBATCH*256*4);
    float*     hmlp  = (float*)    carve((size_t)NBATCH*256*4);

    const int cins[3]  = {7, 64, 128};
    const int outsv[3] = {64, 128, 256};

    k_init_ident<<<1, 256, 0, stream>>>(id1h, id0h);
    k_mask<<<NBATCH, NPT, 0, stream>>>(x_in, p_in, xbuf0, pbuf, maskf, cnts);
    k_stats_f32<<<7, 256, 0, stream>>>(xbuf0, NPT, (long)7*NPT, NPT,
                                       (long)NBATCH*NPT, sums, sumsq);
    k_bn_finalize<<<1, 7, 0, stream>>>(sums, sumsq, fts_g, fts_b,
                                       1.f/((float)NBATCH*NPT),
                                       s_hs, s_hb, s_hsh, s_hbh);
    k_apply_bn_mask<<<(NBATCH*7*NPT)/256, 256, 0, stream>>>(xbuf0, s_hs, s_hb, maskf, 7);

    float* xcur = xbuf0;
    float* xnxt = xbuf1;
    for (int i = 0; i < 3; ++i) {
        int cin = cins[i], O = outsv[i];
        const float* pts = (i == 0) ? pbuf : xcur;
        int D = (i == 0) ? 2 : cin;
        k_knn<<<NBATCH*NPT/8, 256, 0, stream>>>(pts, D, maskf, idxb);

        int C0 = 2*cin, C0pad = (C0 + 31) & ~31;
        k_pack_graph<<<MTOT/256, 256, 0, stream>>>(xcur, cin, C0pad, idxb, P0);

        int otiles = O >> 4, ogroups = O >> 6;
        int ns0 = C0pad >> 5, nsO = O >> 5;
        // layer 0: graph features (identity BN) -> P1
        k_pack_w<<<(otiles*ns0*32 + 255)/256, 256, 0, stream>>>(Wc[i][0], C0, ns0, O, Af);
        k_wmma_gemm<<<ogroups*(MTOT/128), 256, 0, stream>>>(
            Af, ns0, P0, (const h2v*)id1h, (const h2v*)id0h, P1, O, (long)MTOT);
        k_stats_pair<<<O/2, 256, 0, stream>>>(P1, (long)MTOT, sums, sumsq);
        k_bn_finalize<<<1, O, 0, stream>>>(sums, sumsq, bng[i][0], bnb[i][0],
                                           1.f/(float)MTOT, s_hs, s_hb, s_hsh, s_hbh);
        // layer 1: BN+ReLU folded into B load -> P0
        k_pack_w<<<(otiles*nsO*32 + 255)/256, 256, 0, stream>>>(Wc[i][1], O, nsO, O, Af);
        k_wmma_gemm<<<ogroups*(MTOT/128), 256, 0, stream>>>(
            Af, nsO, P1, (const h2v*)s_hsh, (const h2v*)s_hbh, P0, O, (long)MTOT);
        k_stats_pair<<<O/2, 256, 0, stream>>>(P0, (long)MTOT, sums, sumsq);
        k_bn_finalize<<<1, O, 0, stream>>>(sums, sumsq, bng[i][1], bnb[i][1],
                                           1.f/(float)MTOT, s_hs, s_hb, s_hsh, s_hbh);
        // layer 2 -> P1
        k_pack_w<<<(otiles*nsO*32 + 255)/256, 256, 0, stream>>>(Wc[i][2], O, nsO, O, Af);
        k_wmma_gemm<<<ogroups*(MTOT/128), 256, 0, stream>>>(
            Af, nsO, P0, (const h2v*)s_hsh, (const h2v*)s_hbh, P1, O, (long)MTOT);
        k_stats_pair<<<O/2, 256, 0, stream>>>(P1, (long)MTOT, sums, sumsq);
        k_bn_finalize<<<1, O, 0, stream>>>(sums, sumsq, bng[i][2], bnb[i][2],
                                           1.f/(float)MTOT, s_hs, s_hb, s_hsh, s_hbh);
        // shortcut GEMM (pre-masked x, no BN) + its BN
        int cpadsc = (cin + 31) & ~31, nssc = cpadsc >> 5;
        k_pack_xsc<<<M2TOT/256, 256, 0, stream>>>(xcur, cin, cpadsc, Hsc);
        k_pack_w<<<(otiles*nssc*32 + 255)/256, 256, 0, stream>>>(scw[i], cin, nssc, O, Af);
        k_wmma_gemm_f32out<<<ogroups*(M2TOT/128), 256, 0, stream>>>(
            Af, nssc, Hsc, scb, O, (long)M2TOT);
        k_stats_f32<<<O, 256, 0, stream>>>(scb, M2TOT, 0, M2TOT,
                                           (long)M2TOT, sums, sumsq);
        k_bn_finalize<<<1, O, 0, stream>>>(sums, sumsq, scbng[i], scbnb[i],
                                           1.f/(float)M2TOT, s_ss, s_sb, s_ssh, s_sbh);
        // fuse: relu(bn(sc) + mean_k relu(bn(h))) * mask -> next features
        k_epilogue<<<(int)(((long)NBATCH*(O/2)*NPT)/256), 256, 0, stream>>>(
            P1, s_hs, s_hb, scb, s_ss, s_sb, maskf, xnxt, O);
        float* tmp = xcur; xcur = xnxt; xnxt = tmp;
    }

    k_pool<<<NBATCH, 256, 0, stream>>>(xcur, cnts, xmp);
    k_mlp1<<<NBATCH, 256, 0, stream>>>(xmp, W1, b1, hmlp);
    k_mlp2<<<1, 256, 0, stream>>>(hmlp, W2, b2, (float*)d_out);
}
